// MixtureOfExperts_63479616635182
// MI455X (gfx1250) — matrix operand order repaired and checked by reading, not compiled, not run on hardware
//
#include <hip/hip_runtime.h>
#include <stdint.h>

#define N_TOKENS 524288
#define DIM 128
#define NEXP 8

typedef __attribute__((ext_vector_type(16))) __bf16 v16bf;
typedef __attribute__((ext_vector_type(8)))  float  v8f;

struct U32x8 { uint4 a, b; };

static __device__ __forceinline__ v16bf frag16(uint4 a, uint4 b) {
  U32x8 s{a, b};
  return __builtin_bit_cast(v16bf, s);
}

// ---------------- kernel 0: zero counts + cursors ----------------
__global__ void k_zero(int* ctrl) {
  if (threadIdx.x < 16) ctrl[threadIdx.x] = 0;
}

// ---------------- kernel 1: split W into bf16 hi + lo residual ----------------
__global__ void k_wconv(const float* __restrict__ W,
                        unsigned short* __restrict__ Wh,
                        unsigned short* __restrict__ Wl) {
  int i = blockIdx.x * blockDim.x + threadIdx.x;
  float w = W[i];
  __bf16 h = (__bf16)w;
  float r = w - (float)h;
  Wh[i] = __builtin_bit_cast(unsigned short, h);
  Wl[i] = __builtin_bit_cast(unsigned short, (__bf16)r);
}

// ---------------- kernel 2: gating (one wave per token) ----------------
__global__ void __launch_bounds__(256) k_gate(
    const float* __restrict__ x, const float* __restrict__ gW,
    const float* __restrict__ gb, const float* __restrict__ b,
    float* __restrict__ out, int* __restrict__ counts,
    int* __restrict__ top2idx, float2* __restrict__ top2gv)
{
  __shared__ float sGW[NEXP * DIM];
  __shared__ float sB[NEXP * DIM];
  __shared__ float sGb[NEXP];
  for (int i = threadIdx.x; i < NEXP * DIM; i += 256) {
    sGW[i] = gW[i];
    sB[i]  = b[i];
  }
  if (threadIdx.x < NEXP) sGb[threadIdx.x] = gb[threadIdx.x];
  __syncthreads();

  const int wave = threadIdx.x >> 5;
  const int lane = threadIdx.x & 31;
  const int t = blockIdx.x * 8 + wave;

  const float4 xv = *(const float4*)(x + (size_t)t * DIM + lane * 4);

  float s[NEXP];
#pragma unroll
  for (int e = 0; e < NEXP; ++e) {
    const float* w = &sGW[e * DIM + lane * 4];
    s[e] = xv.x * w[0] + xv.y * w[1] + xv.z * w[2] + xv.w * w[3];
  }
  // butterfly reduction: every lane ends with the full dot products
#pragma unroll
  for (int m = 16; m >= 1; m >>= 1)
#pragma unroll
    for (int e = 0; e < NEXP; ++e) s[e] += __shfl_xor(s[e], m, 32);

  float mx = -1e30f;
#pragma unroll
  for (int e = 0; e < NEXP; ++e) { s[e] += sGb[e]; mx = fmaxf(mx, s[e]); }
  float den = 0.f;
  float g[NEXP];
#pragma unroll
  for (int e = 0; e < NEXP; ++e) { g[e] = expf(s[e] - mx); den += g[e]; }
  float inv = 1.f / den;
#pragma unroll
  for (int e = 0; e < NEXP; ++e) g[e] *= inv;

  // bias term: out[t,:] = sum_e g[e] * b[e,:]   (coalesced float4 store)
  float4 o = make_float4(0.f, 0.f, 0.f, 0.f);
#pragma unroll
  for (int e = 0; e < NEXP; ++e) {
    const float* bb = &sB[e * DIM + lane * 4];
    o.x += g[e] * bb[0]; o.y += g[e] * bb[1];
    o.z += g[e] * bb[2]; o.w += g[e] * bb[3];
  }
  *(float4*)(out + (size_t)t * DIM + lane * 4) = o;

  // top-2, earliest index wins ties (matches lax.top_k)
  int e0 = 0;
#pragma unroll
  for (int e = 1; e < NEXP; ++e) if (g[e] > g[e0]) e0 = e;
  int e1 = (e0 == 0) ? 1 : 0;
#pragma unroll
  for (int e = 0; e < NEXP; ++e) if (e != e0 && g[e] > g[e1]) e1 = e;

  if (lane == 0) {
    atomicAdd(&counts[e0], 1);
    atomicAdd(&counts[e1], 1);
    top2idx[t] = e0 | (e1 << 8);
    top2gv[t] = make_float2(g[e0], g[e1]);
  }
}

// ---------------- kernel 3: exclusive prefix over 8 expert counts ----------------
__global__ void k_prefix(int* ctrl) {
  if (threadIdx.x == 0) {
    int acc = 0;
    for (int e = 0; e < NEXP; ++e) { ctrl[16 + e] = acc; acc += ctrl[e]; }
  }
}

// ---------------- kernel 4: scatter (token, gate) into expert buckets ----------------
__global__ void k_scatter(const int* __restrict__ top2idx,
                          const float2* __restrict__ top2gv,
                          int* __restrict__ ctrl,
                          int* __restrict__ btok, float* __restrict__ bgv)
{
  int t = blockIdx.x * blockDim.x + threadIdx.x;
  int pk = top2idx[t];
  float2 gv = top2gv[t];
  int e0 = pk & 255, e1 = (pk >> 8) & 255;
  int s0 = ctrl[16 + e0] + atomicAdd(&ctrl[8 + e0], 1);
  btok[s0] = t; bgv[s0] = gv.x;
  int s1 = ctrl[16 + e1] + atomicAdd(&ctrl[8 + e1], 1);
  btok[s1] = t; bgv[s1] = gv.y;
}

// ---------------- kernel 5: per-expert GEMM, split-bf16 WMMA ----------------
// Block: 256 threads (8 waves). Tile: 128 bucket rows x 128 features.
// W hi/lo staged in LDS with 16B-chunk XOR swizzle -> conflict-free B frag loads.
// ctrl: [0..7]=counts [8..15]=cursors [16..23]=offsets
__global__ void __launch_bounds__(256) k_moe_gemm(
    const float* __restrict__ x,
    const unsigned short* __restrict__ WhG,
    const unsigned short* __restrict__ WlG,
    const int* __restrict__ ctrl,
    const int* __restrict__ btok, const float* __restrict__ bgv,
    float* __restrict__ out)
{
  const int e = blockIdx.y;
  const int cnt = ctrl[e];
  const int rowbase = blockIdx.x * 128;
  if (rowbase >= cnt) return;
  const int off = ctrl[16 + e];

  // 128 rows x 16 chunks of 16B per array (32 KB each).
  // Chunk c of row r stored at physical slot (c ^ (r & 15)).
  __shared__ uint4 sWh4[DIM * 16];
  __shared__ uint4 sWl4[DIM * 16];
  {
    const uint4* srcH = (const uint4*)(WhG + (size_t)e * DIM * DIM);
    const uint4* srcL = (const uint4*)(WlG + (size_t)e * DIM * DIM);
#pragma unroll
    for (int k = 0; k < 8; ++k) {
      int i = threadIdx.x + k * 256;          // linear source chunk
      int row = i >> 4;
      int dst = (i & ~15) | ((i & 15) ^ (row & 15));
      sWh4[dst] = srcH[i];
      sWl4[dst] = srcL[i];
    }
  }
  __syncthreads();

  const int wave = threadIdx.x >> 5;
  const int lane = threadIdx.x & 31;
  const int laneHalf = lane >> 4;   // 0: low 16 lanes, 1: high 16 lanes
  const int lanelo = lane & 15;

  // A-fragment row owned by this lane (same row in both lane halves)
  const int arow = rowbase + wave * 16 + lanelo;
  const int atok = (arow < cnt) ? btok[off + arow] : 0;
  const float agv = (arow < cnt) ? bgv[off + arow] : 0.f;   // 0 => zero A row
  const float* xrow = x + (size_t)atok * DIM;

  v8f acc[8] = {};

#pragma unroll
  for (int ks = 0; ks < 4; ++ks) {
    // A k order chosen here (contiguous, not the ISA table order): low lanes hold K = k0..k0+15,
    // high lanes hold K = k0+16..k0+31 (matches B).
    const int kA = ks * 32 + laneHalf * 16;
    float4 p0 = *(const float4*)(xrow + kA);
    float4 p1 = *(const float4*)(xrow + kA + 4);
    float4 p2 = *(const float4*)(xrow + kA + 8);
    float4 p3 = *(const float4*)(xrow + kA + 12);
    float va[16] = {p0.x,p0.y,p0.z,p0.w, p1.x,p1.y,p1.z,p1.w,
                    p2.x,p2.y,p2.z,p2.w, p3.x,p3.y,p3.z,p3.w};
    v16bf ahi, alo;
#pragma unroll
    for (int i = 0; i < 16; ++i) {
      float v = va[i] * agv;
      __bf16 h = (__bf16)v;
      ahi[i] = h;
      alo[i] = (__bf16)(v - (float)h);
    }

    // B k order chosen here, same as A: lane n = column, low lanes K=k0..k0+15,
    // high lanes K=k0+16..k0+31 (two 16B chunks per lane, XOR-swizzled).
    const int kB = ks * 32 + laneHalf * 16;
    const int c0 = kB >> 3;                 // even logical chunk index
#pragma unroll
    for (int ft = 0; ft < 8; ++ft) {
      const int fr = ft * 16 + lanelo;      // B row (feature), fr & 15 == lanelo
      const int base = fr << 4;
      const int pc0 = c0 ^ lanelo;          // physical chunk of logical c0
      v16bf bhi = frag16(sWh4[base + pc0], sWh4[base + (pc0 ^ 1)]);
      v16bf blo = frag16(sWl4[base + pc0], sWl4[base + (pc0 ^ 1)]);
      acc[ft] = __builtin_amdgcn_wmma_f32_16x16x32_bf16(false, ahi, false, bhi,
                                                        (short)0, acc[ft], false, false);
      acc[ft] = __builtin_amdgcn_wmma_f32_16x16x32_bf16(false, ahi, false, blo,
                                                        (short)0, acc[ft], false, false);
      acc[ft] = __builtin_amdgcn_wmma_f32_16x16x32_bf16(false, alo, false, bhi,
                                                        (short)0, acc[ft], false, false);
    }
  }

  // Epilogue: C layout — VGPR j, low lanes -> M=j, high lanes -> M=j+8; N=lane&15.
  int tks[8];
#pragma unroll
  for (int j = 0; j < 8; ++j) {
    int r = rowbase + wave * 16 + laneHalf * 8 + j;
    tks[j] = (r < cnt) ? btok[off + r] : -1;
  }
#pragma unroll
  for (int ft = 0; ft < 8; ++ft) {
#pragma unroll
    for (int j = 0; j < 8; ++j) {
      if (tks[j] >= 0)
        atomicAdd(out + (size_t)tks[j] * DIM + ft * 16 + lanelo, acc[ft][j]);
    }
  }
}

extern "C" void kernel_launch(void* const* d_in, const int* in_sizes, int n_in,
                              void* d_out, int out_size, void* d_ws, size_t ws_size,
                              hipStream_t stream)
{
  (void)in_sizes; (void)n_in; (void)out_size; (void)ws_size;

  const float* x  = (const float*)d_in[0];
  const float* gW = (const float*)d_in[1];
  const float* gb = (const float*)d_in[2];
  const float* W  = (const float*)d_in[3];
  const float* b  = (const float*)d_in[4];
  float* out = (float*)d_out;

  // workspace layout (~14.5 MiB)
  char* p = (char*)d_ws;
  int* ctrl = (int*)p;                       p += 256;
  int* top2idx = (int*)p;                    p += (size_t)N_TOKENS * 4;
  float2* top2gv = (float2*)p;               p += (size_t)N_TOKENS * 8;
  int* btok = (int*)p;                       p += (size_t)2 * N_TOKENS * 4;
  float* bgv = (float*)p;                    p += (size_t)2 * N_TOKENS * 4;
  unsigned short* Wh = (unsigned short*)p;   p += (size_t)NEXP * DIM * DIM * 2;
  unsigned short* Wl = (unsigned short*)p;

  k_zero<<<1, 32, 0, stream>>>(ctrl);
  k_wconv<<<(NEXP * DIM * DIM) / 256, 256, 0, stream>>>(W, Wh, Wl);
  k_gate<<<N_TOKENS / 8, 256, 0, stream>>>(x, gW, gb, b, out, ctrl, top2idx, top2gv);
  k_prefix<<<1, 32, 0, stream>>>(ctrl);
  k_scatter<<<N_TOKENS / 256, 256, 0, stream>>>(top2idx, top2gv, ctrl, btok, bgv);
  k_moe_gemm<<<dim3(N_TOKENS / 128, NEXP), 256, 0, stream>>>(x, Wh, Wl, ctrl, btok, bgv, out);
}
